// WaveNet_41283225649616
// MI455X (gfx1250) — matrix operand order repaired and checked by reading, not compiled, not run on hardware
//
#include <hip/hip_runtime.h>
#include <hip/hip_bf16.h>
#include <math.h>

// ---------------------------------------------------------------------------
// WaveNet forward for MI455X (gfx1250, wave32, WMMA bf16).
// All big GEMMs run on v_wmma_f32_16x16x32_bf16 with f32 accumulation.
// LDS tiles are stored in fragment-ready (swizzled) layout so each wave
// fragment is a single contiguous 32B-per-lane read (ds_load_b128 x2).
// ---------------------------------------------------------------------------

typedef __attribute__((ext_vector_type(16))) __bf16 v16bf;
typedef __attribute__((ext_vector_type(2)))  __bf16 v2bf;
typedef __attribute__((ext_vector_type(8)))  float  v8f;

#define TLEN 1016
#define CH   2048

__device__ inline v8f vzero8() {
    v8f z = {0.f,0.f,0.f,0.f,0.f,0.f,0.f,0.f};
    return z;
}

// ---- fragment layout helpers (ISA 7.12.2) ---------------------------------
// A fragment (16x32, 16-bit): element e of lane (m, h): k = ((e>>3)<<4) +
// (((e>>1)&3)<<1) + (e&1) + 8h.  Inverse: lane = ((k>>3)&1)*16 + m,
// e = (k>>4)*8 + (k&7).
// B fragment (32x16): element e of lane (n, h): same k order as A, k = a_kidx(e, h).
// Swizzled tile = v16bf[32]: fragment for lane l is element l.

__device__ inline int a_kidx(int e, int half) {
    return ((e >> 3) << 4) + (((e >> 1) & 3) << 1) + (e & 1) + (half << 3);
}

// paired store of (k, k+1) [k even] for row m into an A-swizzled tile
__device__ inline void a_store2(__bf16* Asw, int m, int k, __bf16 lo, __bf16 hi) {
    int h = (k >> 3) & 1;
    int e = ((k >> 4) << 3) + (k & 7);
    v2bf pv; pv[0] = lo; pv[1] = hi;
    *(v2bf*)(Asw + (((h << 4) + m) << 4) + e) = pv;
}

// paired store of (k, k+1) [k even] for column n into a B-swizzled tile
__device__ inline void b_store2(__bf16* Bsw, int k, int n, __bf16 lo, __bf16 hi) {
    int h = (k >> 3) & 1;
    int e = ((k >> 4) << 3) + (k & 7);
    v2bf pv; pv[0] = lo; pv[1] = hi;
    *(v2bf*)(Bsw + (((h << 4) + n) << 4) + e) = pv;
}

// A fragment direct from a global fp32 row-major matrix (small, L2-resident).
__device__ inline v16bf afrag_gf32(const float* W, int rowBase, int ld, int k0, int lane) {
    int m = lane & 15, h = lane >> 4;
    v16bf a;
#pragma unroll
    for (int e = 0; e < 16; ++e)
        a[e] = (__bf16)W[(size_t)(rowBase + m) * ld + k0 + a_kidx(e, h)];
    return a;
}

#define WMMA_BF16(A, B, C) \
    __builtin_amdgcn_wmma_f32_16x16x32_bf16(false, (A), false, (B), (short)0, (C), false, false)

// ---------------------------------------------------------------------------
// mu-law encode
// ---------------------------------------------------------------------------
__global__ void k_mulaw(const float* __restrict__ audio, int* __restrict__ q,
                        int* __restrict__ qout) {
    int t = blockIdx.x * 256 + threadIdx.x;
    if (t < TLEN) {
        float x = fminf(fmaxf(audio[t], -1.f), 1.f);
        float sgn = (x > 0.f) ? 1.f : ((x < 0.f) ? -1.f : 0.f);
        float m = sgn * log1pf(255.f * fabsf(x)) / log1pf(255.f);
        float v = (m + 1.f) * 0.5f * 255.f + 0.5f;
        v = fminf(fmaxf(v, 0.f), 255.f);
        int qi = (int)v;
        q[t] = qi;
        qout[t] = qi;
    }
}

// ---------------------------------------------------------------------------
// Embedding gather: x0[c*T + t] = embed[q[t]*64 + c]
// ---------------------------------------------------------------------------
__global__ void k_embed(const int* __restrict__ q, const float* __restrict__ embed,
                        float* __restrict__ x0) {
    int idx = blockIdx.x * 256 + threadIdx.x;
    if (idx < 64 * TLEN) {
        int c = idx / TLEN, t = idx % TLEN;
        x0[idx] = embed[q[t] * 64 + c];
    }
}

// ---------------------------------------------------------------------------
// Upsampling transposed conv as 8 phase-GEMMs (transposed output):
//   cond[o, 8a+b] = sum_i w_up[i,o,8+b]*F[i,a] + w_up[i,o,b]*F[i,a+1] + b_up[o]
// Block = one 16-wide o tile, all a, all phases; w_up streamed exactly once
// with fully-consumed 64B lines.  Wave w = phase b, 8 a-tile accumulators.
// ---------------------------------------------------------------------------
__global__ __launch_bounds__(256) void k_upsample(const float* __restrict__ feat,
                                                  const float* __restrict__ w_up,
                                                  const float* __restrict__ b_up,
                                                  float* __restrict__ cond1) {
    __shared__ v16bf WswV[16 * 32];   // per-tap B tiles (k=i, n=o)
    __shared__ v16bf A0swV[8 * 32];   // a-tile j  (rows 16j+m)
    __shared__ v16bf A1swV[8 * 32];   // a-tile j shifted (rows 16j+1+m)
    __bf16* Wsw  = (__bf16*)WswV;
    __bf16* A0sw = (__bf16*)A0swV;
    __bf16* A1sw = (__bf16*)A1swV;

    const int tid = threadIdx.x;
    const int lane = tid & 31;
    const int wv = tid >> 5;           // phase b
    const int o0 = blockIdx.x * 16;

    v8f acc[8];
#pragma unroll
    for (int j = 0; j < 8; ++j) acc[j] = vzero8();

    for (int k0 = 0; k0 < CH; k0 += 32) {
        // ---- stage w_up: rows (i0, i0+1) x col oo x 8 taps per unit --------
#pragma unroll
        for (int p = 0; p < 2; ++p) {
            int u = tid * 2 + p;                 // 0..511
            int tg = u & 1;                      // tap group (8 taps)
            int oo = (u >> 1) & 15;
            int i0 = (u >> 5) << 1;              // even i
            const float* r0 = w_up + ((size_t)(k0 + i0) * CH + (o0 + oo)) * 16 + 8 * tg;
            const float* r1 = r0 + (size_t)CH * 16;
            float4 a0 = ((const float4*)r0)[0], a1 = ((const float4*)r0)[1];
            float4 b0 = ((const float4*)r1)[0], b1 = ((const float4*)r1)[1];
            float ta[8] = {a0.x,a0.y,a0.z,a0.w,a1.x,a1.y,a1.z,a1.w};
            float tb[8] = {b0.x,b0.y,b0.z,b0.w,b1.x,b1.y,b1.z,b1.w};
#pragma unroll
            for (int j = 0; j < 8; ++j) {
                int tap = 8 * tg + j;
                b_store2(Wsw + tap * 512, i0, oo, (__bf16)ta[j], (__bf16)tb[j]);
            }
        }
        if (k0 + 32 < CH)
            __builtin_prefetch(w_up + ((size_t)(k0 + 32 + (tid >> 4)) * CH + o0) * 16, 0, 1);

        // ---- stage features: (row aa, k-pair) -> A0 tile and shifted A1 ----
#pragma unroll
        for (int it = 0; it < 8; ++it) {
            int u = it * 256 + tid;              // 0..2047
            int aa = u & 127;
            int kp = (u >> 7) << 1;              // even k
            __bf16 f0 = (__bf16)feat[(k0 + kp) * 128 + aa];
            __bf16 f1 = (__bf16)feat[(k0 + kp + 1) * 128 + aa];
            a_store2(A0sw + (aa >> 4) * 512, aa & 15, kp, f0, f1);
            if (aa >= 1)
                a_store2(A1sw + ((aa - 1) >> 4) * 512, (aa - 1) & 15, kp, f0, f1);
        }
        __syncthreads();

        v16bf Bhi = WswV[(8 + wv) * 32 + lane];  // tap 8+b <-> F[:,a]
        v16bf Blo = WswV[wv * 32 + lane];        // tap   b <-> F[:,a+1]
#pragma unroll
        for (int j = 0; j < 8; ++j) {
            v16bf A0 = A0swV[j * 32 + lane];
            v16bf A1 = A1swV[j * 32 + lane];
            acc[j] = WMMA_BF16(A0, Bhi, acc[j]);
            acc[j] = WMMA_BF16(A1, Blo, acc[j]);
        }
        __syncthreads();
    }

    int n = lane & 15, half = lane >> 4;
    int o = o0 + n;
    float bias = b_up[o];
#pragma unroll
    for (int j = 0; j < 8; ++j) {
#pragma unroll
        for (int v = 0; v < 8; ++v) {
            int a = 16 * j + v + 8 * half;
            if (a < 127)
                cond1[(size_t)o * TLEN + 8 * a + wv] = acc[j][v] + bias;
        }
    }
}

// ---------------------------------------------------------------------------
// Y[2048 x 1016] = W[2048x2048] * X + bias.  64x64 block tiles, K step 32.
// ---------------------------------------------------------------------------
__global__ __launch_bounds__(256) void k_gemm2048(const float* __restrict__ W,
                                                  const float* __restrict__ bias,
                                                  const float* __restrict__ X,
                                                  float* __restrict__ Y) {
    __shared__ v16bf AswV[4 * 32];    // 4 M-tiles
    __shared__ v16bf BswV[4 * 32];    // 4 N-tiles
    __bf16* Asw = (__bf16*)AswV;
    __bf16* Bsw = (__bf16*)BswV;

    int tid = threadIdx.x, lane = tid & 31, wv = tid >> 5;
    int m0 = blockIdx.x * 64, n0 = blockIdx.y * 64;
    int wn = wv & 3, wm2 = wv >> 2;

    v8f acc0 = vzero8(), acc1 = vzero8();
    for (int k0 = 0; k0 < CH; k0 += 32) {
        // A tile: 64 rows x 16 k-pairs, contiguous float2 loads
#pragma unroll
        for (int it = 0; it < 4; ++it) {
            int pr = it * 256 + tid;             // 0..1023
            int r = pr >> 4;
            int kp = (pr & 15) << 1;
            float2 w2 = *(const float2*)&W[(size_t)(m0 + r) * CH + k0 + kp];
            a_store2(Asw + (r >> 4) * 512, r & 15, kp, (__bf16)w2.x, (__bf16)w2.y);
        }
        // B tile: 16 k-pairs x 64 cols, row-pair coalesced loads
#pragma unroll
        for (int it = 0; it < 4; ++it) {
            int pr = it * 256 + tid;
            int col = pr & 63;
            int k = (pr >> 6) << 1;
            int gc = n0 + col;
            float x0 = (gc < TLEN) ? X[(size_t)(k0 + k) * TLEN + gc] : 0.f;
            float x1 = (gc < TLEN) ? X[(size_t)(k0 + k + 1) * TLEN + gc] : 0.f;
            b_store2(Bsw + (col >> 4) * 512, k, col & 15, (__bf16)x0, (__bf16)x1);
        }
        if (k0 + 32 < CH)
            __builtin_prefetch(W + (size_t)(m0 + (tid >> 2)) * CH + k0 + 32, 0, 1);
        __syncthreads();

        v16bf B  = BswV[wn * 32 + lane];
        v16bf A0 = AswV[(2 * wm2) * 32 + lane];
        v16bf A1 = AswV[(2 * wm2 + 1) * 32 + lane];
        acc0 = WMMA_BF16(A0, B, acc0);
        acc1 = WMMA_BF16(A1, B, acc1);
        __syncthreads();
    }

    int n = lane & 15, half = lane >> 4;
    int col = n0 + wn * 16 + n;
    if (col < TLEN) {
#pragma unroll
        for (int v = 0; v < 8; ++v) {
            int r0 = m0 + 32 * wm2 + v + 8 * half;
            Y[(size_t)r0 * TLEN + col] = acc0[v] + bias[r0];
            int r1 = r0 + 16;
            Y[(size_t)r1 * TLEN + col] = acc1[v] + bias[r1];
        }
    }
}

// ---------------------------------------------------------------------------
// One WaveNet layer over a 64-wide time tile.
// Wave w owns c-tile pair (cpair = w&3, cpair+4) and nt-half (w>>2), so the
// tanh and sigmoid halves of ia land in matching register slots and the gate
// is computed entirely in accumulators — no f32 LDS round trip.
// Weight tiles (w_dil, then w_skip) staged swizzled into one shared buffer.
// ---------------------------------------------------------------------------
__global__ __launch_bounds__(256) void k_layer(const float* __restrict__ x_in,
                                               float* __restrict__ x_out,
                                               const float* __restrict__ cond_l,
                                               const float* __restrict__ wd,
                                               const float* __restrict__ bd,
                                               const float* __restrict__ wr,
                                               const float* __restrict__ br,
                                               const float* __restrict__ wsk,
                                               const float* __restrict__ bsk,
                                               float* __restrict__ skip_out,
                                               int dil, int accum, int do_res) {
    __shared__ v16bf XaswV[2 * 4 * 32];  // [kc][nt][lane] : x[:, t-d]
    __shared__ v16bf XbswV[2 * 4 * 32];  // x[:, t]
    __shared__ v16bf AcswV[2 * 4 * 32];  // gated activations
    __shared__ v16bf WgtV[32 * 32];      // w_dil tiles, then reused for w_skip
    __bf16* Xasw = (__bf16*)XaswV;
    __bf16* Xbsw = (__bf16*)XbswV;
    __bf16* Acsw = (__bf16*)AcswV;
    __bf16* Wgt  = (__bf16*)WgtV;

    int tid = threadIdx.x, lane = tid & 31, wv = tid >> 5;
    int t0 = blockIdx.x * 64;
    int n = lane & 15, half = lane >> 4;

    // ---- stage x tiles: channel-pair (j, j+1) x col ------------------------
#pragma unroll
    for (int it = 0; it < 8; ++it) {
        int u = it * 256 + tid;                  // 0..2047
        int col = u & 63;
        int j = (u >> 6) << 1;                   // even channel
        int t = t0 + col, ta = t - dil;
        float xb0 = (t < TLEN) ? x_in[j * TLEN + t] : 0.f;
        float xb1 = (t < TLEN) ? x_in[(j + 1) * TLEN + t] : 0.f;
        float xa0 = (ta >= 0 && t < TLEN) ? x_in[j * TLEN + ta] : 0.f;
        float xa1 = (ta >= 0 && t < TLEN) ? x_in[(j + 1) * TLEN + ta] : 0.f;
        int kc = j >> 5, krel = j & 31;
        int base = (kc * 4 + (col >> 4)) * 512;
        b_store2(Xbsw + base, krel, col & 15, (__bf16)xb0, (__bf16)xb1);
        b_store2(Xasw + base, krel, col & 15, (__bf16)xa0, (__bf16)xa1);
    }
    // ---- stage w_dil: float4 = (k,tap0),(k,tap1),(k+1,tap0),(k+1,tap1) -----
    // tile index = (mt*2 + kc)*2 + tap
#pragma unroll
    for (int it = 0; it < 16; ++it) {
        int u = it * 256 + tid;                  // 0..4095
        int kp = (u & 31) << 1;                  // even k 0..62
        int c  = u >> 5;                         // 0..127
        float4 w4 = *(const float4*)&wd[((size_t)c * 64 + kp) * 2];
        int mt = c >> 4, m = c & 15, kc = kp >> 5, krel = kp & 31;
        a_store2(Wgt + (((mt * 2 + kc) * 2) + 0) * 512, m, krel, (__bf16)w4.x, (__bf16)w4.z);
        a_store2(Wgt + (((mt * 2 + kc) * 2) + 1) * 512, m, krel, (__bf16)w4.y, (__bf16)w4.w);
    }
    __syncthreads();

    // ---- ia GEMM + register-resident gating --------------------------------
    {
        int cpair = wv & 3;                      // lo c-tile; hi = cpair+4
        int nth = wv >> 2;                       // nt = 2*nth + q
        v8f accLo[2], accHi[2];
        accLo[0] = vzero8(); accLo[1] = vzero8();
        accHi[0] = vzero8(); accHi[1] = vzero8();
#pragma unroll
        for (int kc = 0; kc < 2; ++kc) {
            v16bf A0lo = WgtV[(((cpair * 2 + kc) * 2) + 0) * 32 + lane];
            v16bf A1lo = WgtV[(((cpair * 2 + kc) * 2) + 1) * 32 + lane];
            v16bf A0hi = WgtV[((((cpair + 4) * 2 + kc) * 2) + 0) * 32 + lane];
            v16bf A1hi = WgtV[((((cpair + 4) * 2 + kc) * 2) + 1) * 32 + lane];
#pragma unroll
            for (int q = 0; q < 2; ++q) {
                int nt = 2 * nth + q;
                v16bf Ba = XaswV[(kc * 4 + nt) * 32 + lane];
                v16bf Bb = XbswV[(kc * 4 + nt) * 32 + lane];
                accLo[q] = WMMA_BF16(A0lo, Ba, accLo[q]);   // tap0 <-> x[t-d]
                accLo[q] = WMMA_BF16(A1lo, Bb, accLo[q]);   // tap1 <-> x[t]
                accHi[q] = WMMA_BF16(A0hi, Ba, accHi[q]);
                accHi[q] = WMMA_BF16(A1hi, Bb, accHi[q]);
            }
        }
        // gate: acts = tanh(ia_lo) * sigmoid(ia_hi), store swizzled (v pairs)
#pragma unroll
        for (int q = 0; q < 2; ++q) {
            int nt = 2 * nth + q;
            int t = t0 + 16 * nt + n;
            bool valid = (t < TLEN);
#pragma unroll
            for (int v = 0; v < 8; v += 2) {
                int c0 = 16 * cpair + v + 8 * half;      // even channel
                float zl0 = accLo[q][v]     + bd[c0]
                          + (valid ? cond_l[(size_t)c0 * TLEN + t] : 0.f);
                float zl1 = accLo[q][v + 1] + bd[c0 + 1]
                          + (valid ? cond_l[(size_t)(c0 + 1) * TLEN + t] : 0.f);
                float zh0 = accHi[q][v]     + bd[c0 + 64]
                          + (valid ? cond_l[(size_t)(c0 + 64) * TLEN + t] : 0.f);
                float zh1 = accHi[q][v + 1] + bd[c0 + 65]
                          + (valid ? cond_l[(size_t)(c0 + 65) * TLEN + t] : 0.f);
                float a0 = tanhf(zl0) * (1.f / (1.f + expf(-zh0)));
                float a1 = tanhf(zl1) * (1.f / (1.f + expf(-zh1)));
                b_store2(Acsw + ((c0 >> 5) * 4 + nt) * 512, c0 & 31, n,
                         (__bf16)a0, (__bf16)a1);
            }
        }
    }
    __syncthreads();   // acts visible; w_dil reads done -> Wgt reusable

    // ---- restage Wgt with w_skip (256 rows x 32 k-pairs = 32 tiles) --------
#pragma unroll
    for (int it = 0; it < 32; ++it) {
        int u = it * 256 + tid;                  // 0..8191
        int kp = (u & 31) << 1;
        int s = u >> 5;                          // 0..255
        float2 w2 = *(const float2*)&wsk[(size_t)s * 64 + kp];
        a_store2(Wgt + ((s >> 4) * 2 + (kp >> 5)) * 512, s & 15, kp & 31,
                 (__bf16)w2.x, (__bf16)w2.y);
    }
    __syncthreads();

    // ---- skip GEMM: M=256 (wave handles mt = 2w, 2w+1), N=64, K=64 ---------
    {
        v8f acc[2][4];
#pragma unroll
        for (int p = 0; p < 2; ++p)
#pragma unroll
            for (int nt = 0; nt < 4; ++nt) acc[p][nt] = vzero8();
#pragma unroll
        for (int kc = 0; kc < 2; ++kc) {
            v16bf Bn[4];
#pragma unroll
            for (int nt = 0; nt < 4; ++nt)
                Bn[nt] = AcswV[(kc * 4 + nt) * 32 + lane];
#pragma unroll
            for (int p = 0; p < 2; ++p) {
                v16bf A = WgtV[((2 * wv + p) * 2 + kc) * 32 + lane];
#pragma unroll
                for (int nt = 0; nt < 4; ++nt)
                    acc[p][nt] = WMMA_BF16(A, Bn[nt], acc[p][nt]);
            }
        }
#pragma unroll
        for (int p = 0; p < 2; ++p) {
#pragma unroll
            for (int nt = 0; nt < 4; ++nt) {
#pragma unroll
                for (int v = 0; v < 8; ++v) {
                    int s = 16 * (2 * wv + p) + v + 8 * half;
                    int t = t0 + 16 * nt + n;
                    if (t < TLEN) {
                        float val = acc[p][nt][v] + bsk[s];
                        size_t off = (size_t)s * TLEN + t;
                        skip_out[off] = accum ? (skip_out[off] + val) : val;
                    }
                }
            }
        }
    }

    // ---- residual GEMM: 16 tiles, wave handles tiles 2w, 2w+1 --------------
    if (do_res) {
        v8f acc[2];
        acc[0] = vzero8(); acc[1] = vzero8();
#pragma unroll
        for (int kc = 0; kc < 2; ++kc) {
            int k0 = kc * 32;
#pragma unroll
            for (int qq = 0; qq < 2; ++qq) {
                int tile = 2 * wv + qq;
                int mt = tile >> 2, nt = tile & 3;
                v16bf A = afrag_gf32(wr, 16 * mt, 64, k0, lane);
                v16bf B = AcswV[(kc * 4 + nt) * 32 + lane];
                acc[qq] = WMMA_BF16(A, B, acc[qq]);
            }
        }
#pragma unroll
        for (int qq = 0; qq < 2; ++qq) {
            int tile = 2 * wv + qq;
            int mt = tile >> 2, nt = tile & 3;
#pragma unroll
            for (int v = 0; v < 8; ++v) {
                int r = 16 * mt + v + 8 * half;
                int t = t0 + 16 * nt + n;
                if (t < TLEN)
                    x_out[(size_t)r * TLEN + t] =
                        x_in[(size_t)r * TLEN + t] + acc[qq][v] + br[r];
            }
        }
    }
}

// ---------------------------------------------------------------------------
// Head: out = shift_right( w_end * relu(w_out * relu(skip)) )
// Block = 32-wide time tile; two M=256,K=256 WMMA GEMMs via swizzled LDS.
// ---------------------------------------------------------------------------
__global__ __launch_bounds__(256) void k_head(const float* __restrict__ skip,
                                              const float* __restrict__ w_out,
                                              const float* __restrict__ w_end,
                                              float* __restrict__ out) {
    __shared__ v16bf RswV[8 * 2 * 32];   // [kc][nt][lane] relu(skip)
    __shared__ v16bf HswV[8 * 2 * 32];   // hidden activations
    __bf16* Rsw = (__bf16*)RswV;
    __bf16* Hsw = (__bf16*)HswV;

    int tid = threadIdx.x, lane = tid & 31, wv = tid >> 5;
    int t0 = blockIdx.x * 32;
    int n = lane & 15, half = lane >> 4;

    // stage relu(skip): row-pair (s, s+1) x col
#pragma unroll
    for (int it = 0; it < 16; ++it) {
        int u = it * 256 + tid;                  // 0..4095
        int col = u & 31;
        int s = (u >> 5) << 1;                   // even row
        int gc = t0 + col;
        float r0 = (gc < TLEN) ? fmaxf(skip[(size_t)s * TLEN + gc], 0.f) : 0.f;
        float r1 = (gc < TLEN) ? fmaxf(skip[(size_t)(s + 1) * TLEN + gc], 0.f) : 0.f;
        int kc = s >> 5, krel = s & 31;
        b_store2(Rsw + (kc * 2 + (col >> 4)) * 512, krel, col & 15,
                 (__bf16)r0, (__bf16)r1);
    }
    __syncthreads();

    // GEMM1: h = relu(w_out * R), written directly into swizzled Hsw
    {
        v8f acc[2][2];
#pragma unroll
        for (int p = 0; p < 2; ++p)
#pragma unroll
            for (int nt = 0; nt < 2; ++nt) acc[p][nt] = vzero8();
        for (int k0 = 0; k0 < 256; k0 += 32) {
            v16bf Bn[2];
#pragma unroll
            for (int nt = 0; nt < 2; ++nt)
                Bn[nt] = RswV[((k0 >> 5) * 2 + nt) * 32 + lane];
#pragma unroll
            for (int p = 0; p < 2; ++p) {
                v16bf A = afrag_gf32(w_out, 16 * (2 * wv + p), 256, k0, lane);
#pragma unroll
                for (int nt = 0; nt < 2; ++nt)
                    acc[p][nt] = WMMA_BF16(A, Bn[nt], acc[p][nt]);
            }
        }
#pragma unroll
        for (int p = 0; p < 2; ++p)
#pragma unroll
            for (int nt = 0; nt < 2; ++nt)
#pragma unroll
                for (int v = 0; v < 8; v += 2) {   // row pairs (s even)
                    int s = 16 * (2 * wv + p) + v + 8 * half;
                    float h0 = fmaxf(acc[p][nt][v], 0.f);
                    float h1 = fmaxf(acc[p][nt][v + 1], 0.f);
                    int kc = s >> 5, krel = s & 31;
                    b_store2(Hsw + (kc * 2 + nt) * 512, krel, n,
                             (__bf16)h0, (__bf16)h1);
                }
    }
    __syncthreads();

    // GEMM2: o = w_end * h, then shift right by one sample
    {
        v8f acc[2][2];
#pragma unroll
        for (int p = 0; p < 2; ++p)
#pragma unroll
            for (int nt = 0; nt < 2; ++nt) acc[p][nt] = vzero8();
        for (int k0 = 0; k0 < 256; k0 += 32) {
            v16bf Bn[2];
#pragma unroll
            for (int nt = 0; nt < 2; ++nt)
                Bn[nt] = HswV[((k0 >> 5) * 2 + nt) * 32 + lane];
#pragma unroll
            for (int p = 0; p < 2; ++p) {
                v16bf A = afrag_gf32(w_end, 16 * (2 * wv + p), 256, k0, lane);
#pragma unroll
                for (int nt = 0; nt < 2; ++nt)
                    acc[p][nt] = WMMA_BF16(A, Bn[nt], acc[p][nt]);
            }
        }
#pragma unroll
        for (int p = 0; p < 2; ++p)
#pragma unroll
            for (int nt = 0; nt < 2; ++nt)
#pragma unroll
                for (int v = 0; v < 8; ++v) {
                    int s = 16 * (2 * wv + p) + v + 8 * half;
                    int t = t0 + 16 * nt + n;
                    if (t + 1 < TLEN)
                        out[(size_t)s * TLEN + t + 1] = acc[p][nt][v];
                    if (t == 0)
                        out[(size_t)s * TLEN] = 0.f;
                }
    }
}

// ---------------------------------------------------------------------------
// Host launcher
// ---------------------------------------------------------------------------
extern "C" void kernel_launch(void* const* d_in, const int* in_sizes, int n_in,
                              void* d_out, int out_size, void* d_ws, size_t ws_size,
                              hipStream_t stream) {
    const float* features = (const float*)d_in[0];
    const float* audio    = (const float*)d_in[1];
    const float* w_up     = (const float*)d_in[2];
    const float* b_up     = (const float*)d_in[3];
    const float* w_cond   = (const float*)d_in[4];
    const float* b_cond   = (const float*)d_in[5];
    const float* embed    = (const float*)d_in[6];
    const float* w_dil    = (const float*)d_in[7];
    const float* b_dil    = (const float*)d_in[8];
    const float* w_res    = (const float*)d_in[9];
    const float* b_res    = (const float*)d_in[10];
    const float* w_skip   = (const float*)d_in[11];
    const float* b_skip   = (const float*)d_in[12];
    const float* w_out    = (const float*)d_in[13];
    const float* w_end    = (const float*)d_in[14];

    float* out_f = (float*)d_out;
    int*   q_out = ((int*)d_out) + 256 * TLEN;

    // workspace carve
    float* cond1 = (float*)d_ws;                         // 2048 x 1016 (also cond_acts)
    float* cond2 = cond1 + (size_t)CH * TLEN;            // 2048 x 1016
    float* condA = cond1;                                // reuse: cond1 dead after GEMM1
    float* xa    = cond2 + (size_t)CH * TLEN;            // 64 x 1016
    float* xb    = xa + (size_t)64 * TLEN;               // 64 x 1016
    float* skip  = xb + (size_t)64 * TLEN;               // 256 x 1016
    int*   qbuf  = (int*)(skip + (size_t)256 * TLEN);    // 1016

    k_mulaw<<<4, 256, 0, stream>>>(audio, qbuf, q_out);

    k_upsample<<<128, 256, 0, stream>>>(features, w_up, b_up, cond1);

    dim3 g2(32, 16);
    k_gemm2048<<<g2, 256, 0, stream>>>(w_cond, b_cond, cond1, cond2);
    k_gemm2048<<<g2, 256, 0, stream>>>(w_cond, b_cond, cond2, condA);

    k_embed<<<(64 * TLEN + 255) / 256, 256, 0, stream>>>(qbuf, embed, xa);

    for (int i = 0; i < 16; ++i) {
        int d = 1 << (i % 8);
        const float* xin = (i & 1) ? xb : xa;
        float*       xout = (i & 1) ? xa : xb;
        const float* wr = w_res + (size_t)((i < 15) ? i : 0) * 64 * 64;
        const float* br = b_res + (size_t)((i < 15) ? i : 0) * 64;
        k_layer<<<16, 256, 0, stream>>>(
            xin, xout,
            condA + (size_t)i * 128 * TLEN,
            w_dil + (size_t)i * 128 * 64 * 2, b_dil + (size_t)i * 128,
            wr, br,
            w_skip + (size_t)i * 256 * 64, b_skip + (size_t)i * 256,
            skip, d, (i == 0) ? 0 : 1, (i < 15) ? 1 : 0);
    }

    k_head<<<32, 256, 0, stream>>>(skip, w_out, w_end, out_f);
}
